// ProbAttention_7198365188159
// MI455X (gfx1250) — matrix operand order repaired and checked by reading, not compiled, not run on hardware
//
#include <hip/hip_runtime.h>
#include <hip/hip_bf16.h>
#include <cstdint>
#include <cstddef>

// ---------------- problem constants (from reference) ----------------
#define B_     8
#define L_     2048
#define H_     8
#define D_     64
#define U_     40            // u = top-k count = 5*ceil(log(2048)) = 40
#define UP_    48            // padded to 3 WMMA row tiles of 16
#define SAMPK_ 40            // sample_k
#define NW_    4             // waves per block in attention kernel
#define KB_    32            // keys per flash block (= bf16 WMMA K)
#define ITER_  (L_ / (NW_ * KB_))   // 16 key blocks per wave
#define SCALE_ 0.125f        // 1/sqrt(64)
#define NEG_BIG (-1.0e30f)

typedef __attribute__((ext_vector_type(16))) __bf16 v16bf;
typedef __attribute__((ext_vector_type(8)))  float  v8f;

__device__ __forceinline__ float rowmax16(float v) {  // reduce across 16-lane group
  v = fmaxf(v, __shfl_xor(v, 1, 32));
  v = fmaxf(v, __shfl_xor(v, 2, 32));
  v = fmaxf(v, __shfl_xor(v, 4, 32));
  v = fmaxf(v, __shfl_xor(v, 8, 32));
  return v;
}
__device__ __forceinline__ float rowsum16(float v) {
  v += __shfl_xor(v, 1, 32);
  v += __shfl_xor(v, 2, 32);
  v += __shfl_xor(v, 4, 32);
  v += __shfl_xor(v, 8, 32);
  return v;
}
__device__ __forceinline__ float sum32(float v) {
  v += __shfl_xor(v, 1, 32);
  v += __shfl_xor(v, 2, 32);
  v += __shfl_xor(v, 4, 32);
  v += __shfl_xor(v, 8, 32);
  v += __shfl_xor(v, 16, 32);
  return v;
}

// ============ Kernel 1: sampled sparsity scores M[b,h,l] ============
// One wave per (b,h,l). Lane d holds q[d], q[d+32]; 40 sampled dots.
__global__ __launch_bounds__(256) void k_mscore(
    const float* __restrict__ Q, const float* __restrict__ K,
    const int* __restrict__ idx, float* __restrict__ M) {
  const int gw   = blockIdx.x * 8 + (threadIdx.x >> 5);
  const int lane = threadIdx.x & 31;
  const int l  = gw % L_;
  const int bh = gw / L_;
  const int b  = bh / H_, h = bh % H_;
  const float* qrow = Q + (((size_t)b * L_ + l) * H_ + h) * D_;
  const float q0 = qrow[lane], q1 = qrow[lane + 32];
  float mx = NEG_BIG, sm = 0.f;
  for (int s = 0; s < SAMPK_; ++s) {
    const int ks = idx[l * SAMPK_ + s];
    const float* krow = K + (((size_t)b * L_ + ks) * H_ + h) * D_;
    float dot = sum32(q0 * krow[lane] + q1 * krow[lane + 32]);
    mx = fmaxf(mx, dot);
    sm += dot;
  }
  if (lane == 0) M[(size_t)bh * L_ + l] = mx - sm * (1.0f / (float)L_);
}

// ============ Kernel 2: top-40 per (b,h) (stable, lowest index wins ties) ============
__global__ __launch_bounds__(256) void k_topk(
    const float* __restrict__ M, int* __restrict__ top) {
  const int bh = blockIdx.x;
  __shared__ float sv[L_];
  __shared__ float rv[256];
  __shared__ int   ri[256];
  for (int i = threadIdx.x; i < L_; i += 256) sv[i] = M[(size_t)bh * L_ + i];
  __syncthreads();
  for (int u = 0; u < U_; ++u) {
    float best = NEG_BIG * 2.0f; int bi = 0;
    for (int i = threadIdx.x; i < L_; i += 256) {
      float v = sv[i];
      if (v > best) { best = v; bi = i; }
    }
    rv[threadIdx.x] = best; ri[threadIdx.x] = bi;
    __syncthreads();
    for (int off = 128; off > 0; off >>= 1) {
      if (threadIdx.x < off) {
        float ov = rv[threadIdx.x + off]; int oi = ri[threadIdx.x + off];
        if (ov > rv[threadIdx.x] ||
            (ov == rv[threadIdx.x] && oi < ri[threadIdx.x])) {
          rv[threadIdx.x] = ov; ri[threadIdx.x] = oi;
        }
      }
      __syncthreads();
    }
    if (threadIdx.x == 0) { top[bh * U_ + u] = ri[0]; sv[ri[0]] = NEG_BIG * 2.0f; }
    __syncthreads();
  }
}

// ============ Kernel 3: V mean over L, broadcast-fill output ============
__global__ __launch_bounds__(256) void k_vmean_fill(
    const float* __restrict__ V, float* __restrict__ out) {
  const int bh = blockIdx.x, b = bh / H_, h = bh % H_;
  const int d = threadIdx.x & 63, g = threadIdx.x >> 6;  // 4 l-groups x 64 d
  __shared__ float part[4][D_];
  __shared__ float vmean[D_];
  float s = 0.f;
  for (int l = g; l < L_; l += 4)
    s += V[(((size_t)b * L_ + l) * H_ + h) * D_ + d];
  part[g][d] = s;
  __syncthreads();
  if (threadIdx.x < D_)
    vmean[threadIdx.x] = (part[0][threadIdx.x] + part[1][threadIdx.x] +
                          part[2][threadIdx.x] + part[3][threadIdx.x]) *
                         (1.0f / (float)L_);
  __syncthreads();
  const float vm = vmean[d];
  for (int l = g; l < L_; l += 4)
    out[(((size_t)b * L_ + l) * H_ + h) * D_ + d] = vm;
}

// ============ Kernel 4: flash attention for the 40 selected queries ============
// One block per (b,h); 4 waves each own 512 keys (16 blocks of 32).
// bf16 WMMA 16x16x32, fp32 accum. 24 v_wmma per key block per wave.
__global__ __launch_bounds__(128) void k_attn(
    const float* __restrict__ Q, const float* __restrict__ K,
    const float* __restrict__ V, const int* __restrict__ top,
    float* __restrict__ out) {
  const int bh = blockIdx.x, b = bh / H_, h = bh % H_;
  const int tid = threadIdx.x;
  const int wid = tid >> 5;
  const int lane = tid & 31;
  const int lh  = lane >> 4;   // lane-half: 0 or 1
  const int l16 = lane & 15;

  __shared__ int    sTop[UP_];
  __shared__ __bf16 sP[NW_][UP_][KB_];  // bf16 probabilities, row-major
  __shared__ float  sO[UP_][D_];
  __shared__ float  sMw[NW_][UP_];
  __shared__ float  sMg[UP_], sLsum[UP_];

  if (tid < UP_) {
    sTop[tid]  = (tid < U_) ? top[bh * U_ + tid] : 0;
    sLsum[tid] = 0.f;
  }
  for (int i = tid; i < UP_ * D_; i += 128) (&sO[0][0])[i] = 0.f;
  __syncthreads();

  // ---- stage Q_reduce A-operands in registers (held whole kernel) ----
  // A layout (16x32 bf16): row = l16, K(e) = lh*16 + e
  // 1/sqrt(D) is folded into Q here (native v_cvt bf16 converts).
  v16bf qa[3][2];
#pragma unroll
  for (int rt = 0; rt < 3; ++rt) {
    const int row = rt * 16 + l16;
    const bool valid = row < U_;
    const float* qrow =
        Q + (((size_t)b * L_ + (valid ? sTop[row] : 0)) * H_ + h) * D_;
#pragma unroll
    for (int dc = 0; dc < 2; ++dc) {
      v16bf w;
#pragma unroll
      for (int e = 0; e < 16; ++e) {
        const int d = dc * 32 + lh * 16 + e;
        w[e] = (__bf16)(valid ? qrow[d] * SCALE_ : 0.f);
      }
      qa[rt][dc] = w;
    }
  }

  // ---- flash state ----
  v8f o[3][4];
  float m_run[3][8], l_run[3][8];
#pragma unroll
  for (int rt = 0; rt < 3; ++rt) {
#pragma unroll
    for (int dt = 0; dt < 4; ++dt) o[rt][dt] = (v8f){};
#pragma unroll
    for (int r = 0; r < 8; ++r) { m_run[rt][r] = NEG_BIG; l_run[rt][r] = 0.f; }
  }

  for (int it = 0; it < ITER_; ++it) {
    const int kbase = wid * (ITER_ * KB_) + it * KB_;

    if (it + 1 < ITER_) {  // prefetch next K block (global_prefetch_b8)
      const float* nk =
          K + (((size_t)b * L_ + (kbase + KB_ + l16)) * H_ + h) * D_;
      __builtin_prefetch(nk, 0, 1);
    }

    // ---- load K^T B-operands: B[kk=d][n=key]; B layout K(e)=lh*16+e, N=l16
    v16bf kb[2][2];
#pragma unroll
    for (int nc = 0; nc < 2; ++nc) {
      const int key = kbase + nc * 16 + l16;
      const float* krow = K + (((size_t)b * L_ + key) * H_ + h) * D_;
#pragma unroll
      for (int dc = 0; dc < 2; ++dc) {
        v16bf w;
#pragma unroll
        for (int e = 0; e < 16; ++e) {
          const int d = dc * 32 + lh * 16 + e;
          w[e] = (__bf16)krow[d];
        }
        kb[nc][dc] = w;
      }
    }

    // ---- S = (Q/sqrt(D)) Kt : 12 WMMAs (scale pre-folded into Q) ----
    v8f s[3][2];
#pragma unroll
    for (int rt = 0; rt < 3; ++rt) {
#pragma unroll
      for (int nc = 0; nc < 2; ++nc) {
        v8f c = (v8f){};
        c = __builtin_amdgcn_wmma_f32_16x16x32_bf16(
            false, qa[rt][0], false, kb[nc][0], (short)0, c, false, false);
        c = __builtin_amdgcn_wmma_f32_16x16x32_bf16(
            false, qa[rt][1], false, kb[nc][1], (short)0, c, false, false);
        s[rt][nc] = c;
      }
    }

    // ---- online softmax over the 32-key block ----
    // C layout: element r -> row rt*16 + lh*8 + r, col = l16 (+16 for tile 1)
#pragma unroll
    for (int rt = 0; rt < 3; ++rt) {
#pragma unroll
      for (int r = 0; r < 8; ++r) {
        const float bm = rowmax16(fmaxf(s[rt][0][r], s[rt][1][r]));
        const float m_new = fmaxf(m_run[rt][r], bm);
        const float corr = __expf(m_run[rt][r] - m_new);
        m_run[rt][r] = m_new;
        const float p0 = __expf(s[rt][0][r] - m_new);
        const float p1 = __expf(s[rt][1][r] - m_new);
        s[rt][0][r] = p0; s[rt][1][r] = p1;
        l_run[rt][r] = l_run[rt][r] * corr + rowsum16(p0 + p1);
#pragma unroll
        for (int dt = 0; dt < 4; ++dt) o[rt][dt][r] *= corr;
      }
      // publish P (bf16) to this wave's LDS strip, row-major [row][key]
#pragma unroll
      for (int r = 0; r < 8; ++r) {
        const int row = rt * 16 + lh * 8 + r;
        sP[wid][row][l16]      = (__bf16)s[rt][0][r];
        sP[wid][row][16 + l16] = (__bf16)s[rt][1][r];
      }
    }

    // ---- load V B-operands: B[kk=key][n=d]; element e -> V[kbase+lh*16+e][dt*16+l16]
    v16bf vb[4];
#pragma unroll
    for (int dt = 0; dt < 4; ++dt) {
      v16bf w;
#pragma unroll
      for (int e = 0; e < 16; ++e) {
        w[e] = (__bf16)V[(((size_t)b * L_ + (kbase + lh * 16 + e)) * H_ + h) *
                             D_ +
                         dt * 16 + l16];
      }
      vb[dt] = w;
    }

    // intra-wave LDS handoff of P (writes above, cross-lane reads below)
    asm volatile("s_wait_dscnt 0" ::: "memory");

    // ---- O += P * V : 12 WMMAs ----
#pragma unroll
    for (int rt = 0; rt < 3; ++rt) {
      v16bf pa;
#pragma unroll
      for (int e = 0; e < 16; ++e) {
        const int kk = lh * 16 + e;  // A-layout K index
        pa[e] = sP[wid][rt * 16 + l16][kk];
      }
#pragma unroll
      for (int dt = 0; dt < 4; ++dt)
        o[rt][dt] = __builtin_amdgcn_wmma_f32_16x16x32_bf16(
            false, pa, false, vb[dt], (short)0, o[rt][dt], false, false);
    }
  }

  // ---- merge 4 wave-partials: m, l, O ----
  if (l16 == 0) {
#pragma unroll
    for (int rt = 0; rt < 3; ++rt)
#pragma unroll
      for (int r = 0; r < 8; ++r)
        sMw[wid][rt * 16 + lh * 8 + r] = m_run[rt][r];
  }
  __syncthreads();
  if (tid < UP_) {
    float mg = NEG_BIG;
    for (int w = 0; w < NW_; ++w) mg = fmaxf(mg, sMw[w][tid]);
    sMg[tid] = mg;
  }
  __syncthreads();
  for (int w = 0; w < NW_; ++w) {
    if (wid == w) {
#pragma unroll
      for (int rt = 0; rt < 3; ++rt)
#pragma unroll
        for (int r = 0; r < 8; ++r) {
          const int row = rt * 16 + lh * 8 + r;
          const float corr = __expf(m_run[rt][r] - sMg[row]);
          if (l16 == 0) sLsum[row] += l_run[rt][r] * corr;
#pragma unroll
          for (int dt = 0; dt < 4; ++dt)
            sO[row][dt * 16 + l16] += o[rt][dt][r] * corr;
        }
    }
    __syncthreads();
  }

  // ---- scatter the 40 finished rows into the output ----
  for (int i = tid; i < U_ * D_; i += 128) {
    const int row = i >> 6, d = i & 63;
    const float val = sO[row][d] / sLsum[row];
    out[(((size_t)b * L_ + sTop[row]) * H_ + h) * D_ + d] = val;
  }
}

// ============================ launcher ============================
extern "C" void kernel_launch(void* const* d_in, const int* in_sizes, int n_in,
                              void* d_out, int out_size, void* d_ws,
                              size_t ws_size, hipStream_t stream) {
  (void)in_sizes; (void)n_in; (void)out_size; (void)ws_size;
  const float* Q   = (const float*)d_in[0];
  const float* K   = (const float*)d_in[1];
  const float* V   = (const float*)d_in[2];
  // d_in[3] = attn_mask (unused by the reference math)
  const int*   idx = (const int*)d_in[4];
  float* out = (float*)d_out;

  float* M   = (float*)d_ws;                                        // B*H*L f32
  int*   top = (int*)((char*)d_ws + (size_t)B_ * H_ * L_ * sizeof(float));

  k_mscore<<<(B_ * H_ * L_) / 8, 256, 0, stream>>>(Q, K, idx, M);
  k_topk<<<B_ * H_, 256, 0, stream>>>(M, top);
  k_vmean_fill<<<B_ * H_, 256, 0, stream>>>(V, out);
  k_attn<<<B_ * H_, 128, 0, stream>>>(Q, K, V, top, out);
}
